// FusionBlock_12549894439391
// MI455X (gfx1250) — compile-verified, matrix operand order checked by reading, not run on hardware
//
#include <hip/hip_runtime.h>

// ---------------------------------------------------------------------------
// FusionBlock (VMamba-style cross-scan SSM) for MI455X / gfx1250, wave32.
// GEMMs -> v_wmma_f32_16x16x32_bf16 (fully unrolled K, templated B stride so
// all addresses fold into constant instruction offsets; no bounds checks:
// weights zero-padded in ws so every load is unconditional / vectorizable);
// selective scan -> 1 wave per channel, lane n = state n (DS==32==wave32).
// ---------------------------------------------------------------------------

typedef __bf16 bf16_t;
typedef bf16_t v16bf __attribute__((ext_vector_type(16)));
typedef float  v8f   __attribute__((ext_vector_type(8)));

// Problem constants (match reference)
constexpr int HH   = 48;
constexpr int WW   = 48;
constexpr int DM   = 96;
constexpr int DS   = 32;
constexpr int DI   = 144;
constexpr int DRNK = 6;
constexpr int KDIR = 4;
constexpr int LTOK = HH * WW;        // 2304
constexpr int L2   = 2 * LTOK;       // 4608
constexpr int KD   = KDIR * DI;      // 576 channels
constexpr int PROJ = DRNK + 2 * DS;  // 70
constexpr int PROJP = 80;            // padded M for x/y_proj
constexpr int KPAD_DI  = 160;        // 144 -> 160
constexpr int KPAD_DT  = 32;         // 6   -> 32

static __device__ __forceinline__ bf16_t f2bf(float f) { return static_cast<bf16_t>(f); }

// ---------------- generic batched bf16 WMMA GEMM ----------------------------
// C[m,n] = sum_k A[m,k]*B[k,n].  NO bounds checks: caller guarantees
//  - M,N multiples of 16 (tilesM*tilesN tiles exactly cover C)
//  - KSTEPS*32 k-range readable on both operands; one operand is ZERO in
//    any k >= true K, so the other may read arbitrary (valid) memory there.
// LDBK is compile-time so strided B loads use constant instruction offsets;
// LDBK==1 vectorizes B exactly like A (float4 -> global_load_b128).
template <int KSTEPS, int LDBK>
__global__ __launch_bounds__(256)
void wmma_gemm_bf16(const float* __restrict__ A, const float* __restrict__ B,
                    float* __restrict__ C, int tilesN, int tileCount,
                    int lda, int ldbN, int ldc,
                    long sA, long sB, long sC) {
  const int batch = blockIdx.y;
  const float* Ab = A + (long)batch * sA;
  const float* Bb = B + (long)batch * sB;
  float*       Cb = C + (long)batch * sC;

  const int lane = threadIdx.x;
  const int tile = blockIdx.x * blockDim.y + threadIdx.y;   // wave-uniform
  if (tile >= tileCount) return;
  const int m0 = (tile / tilesN) * 16;
  const int n0 = (tile % tilesN) * 16;
  const int half = lane >> 4, r = lane & 15;

  const float* arow = Ab + (long)(m0 + r) * lda;   // this lane's A row
  const float* bcol = Bb + (long)(n0 + r) * ldbN;  // this lane's B column

  v8f acc = {};
#pragma unroll
  for (int s = 0; s < KSTEPS; ++s) {
    const int kk = s * 32;
    // A: two runs of 8 consecutive floats -> 4x global_load_b128
    float4 a0 = *(const float4*)(arow + kk + half * 8);
    float4 a1 = *(const float4*)(arow + kk + half * 8 + 4);
    float4 a2 = *(const float4*)(arow + kk + 16 + half * 8);
    float4 a3 = *(const float4*)(arow + kk + 16 + half * 8 + 4);
    v16bf a, b;
    a[0] = f2bf(a0.x); a[1] = f2bf(a0.y); a[2]  = f2bf(a0.z); a[3]  = f2bf(a0.w);
    a[4] = f2bf(a1.x); a[5] = f2bf(a1.y); a[6]  = f2bf(a1.z); a[7]  = f2bf(a1.w);
    a[8] = f2bf(a2.x); a[9] = f2bf(a2.y); a[10] = f2bf(a2.z); a[11] = f2bf(a2.w);
    a[12] = f2bf(a3.x); a[13] = f2bf(a3.y); a[14] = f2bf(a3.z); a[15] = f2bf(a3.w);
    if constexpr (LDBK == 1) {
      float4 b0 = *(const float4*)(bcol + kk + half * 8);
      float4 b1 = *(const float4*)(bcol + kk + half * 8 + 4);
      float4 b2 = *(const float4*)(bcol + kk + 16 + half * 8);
      float4 b3 = *(const float4*)(bcol + kk + 16 + half * 8 + 4);
      b[0] = f2bf(b0.x); b[1] = f2bf(b0.y); b[2]  = f2bf(b0.z); b[3]  = f2bf(b0.w);
      b[4] = f2bf(b1.x); b[5] = f2bf(b1.y); b[6]  = f2bf(b1.z); b[7]  = f2bf(b1.w);
      b[8] = f2bf(b2.x); b[9] = f2bf(b2.y); b[10] = f2bf(b2.z); b[11] = f2bf(b2.w);
      b[12] = f2bf(b3.x); b[13] = f2bf(b3.y); b[14] = f2bf(b3.z); b[15] = f2bf(b3.w);
    } else {
#pragma unroll
      for (int e = 0; e < 8; ++e) {
        // constant byte offsets: (kk + half*8 + e [+16]) * LDBK * 4
        b[e]     = f2bf(bcol[(long)(kk + half * 8 + e) * LDBK]);
        b[e + 8] = f2bf(bcol[(long)(kk + 16 + half * 8 + e) * LDBK]);
      }
    }
    acc = __builtin_amdgcn_wmma_f32_16x16x32_bf16(false, a, false, b,
                                                  (short)0, acc, false, false);
  }

  // C/D: VGPR v -> row m0 + v + 8*half, col n0 + r (unconditional store)
  float* crow = Cb + (long)(m0 + half * 8) * ldc + n0 + r;
#pragma unroll
  for (int v = 0; v < 8; ++v) crow[(long)v * ldc] = acc[v];
}

// ---------------- zero-padded weight copy (tiny) ----------------------------
__global__ __launch_bounds__(256)
void padcopy_kernel(const float* __restrict__ src, float* __restrict__ dst,
                    int rows, int cols, int rowsP, int colsP, int nb) {
  int idx = blockIdx.x * blockDim.x + threadIdx.x;
  if (idx >= nb * rowsP * colsP) return;
  int cp = idx % colsP;
  int t  = idx / colsP;
  int rp = t % rowsP;
  int b  = t / rowsP;
  dst[idx] = (rp < rows && cp < cols) ? src[((long)b * rows + rp) * cols + cp] : 0.0f;
}

// ---------------- depthwise 3x3 conv + bias + SiLU --------------------------
__global__ __launch_bounds__(256)
void conv_silu_kernel(const float* __restrict__ xz, const float* __restrict__ yz,
                      const float* __restrict__ cw, const float* __restrict__ cb,
                      float* __restrict__ cbuf) {
  int idx = blockIdx.x * blockDim.x + threadIdx.x;
  if (idx >= 2 * DI * LTOK) return;
  int br  = idx / (DI * LTOK);
  int rem = idx - br * DI * LTOK;
  int d = rem / LTOK, l = rem % LTOK;
  int h = l / WW, w = l % WW;
  const float* src = br ? yz : xz;
  float s = cb[d];
#pragma unroll
  for (int dy = -1; dy <= 1; ++dy)
#pragma unroll
    for (int dx = -1; dx <= 1; ++dx) {
      int hh = h + dy, ww = w + dx;
      if (hh >= 0 && hh < HH && ww >= 0 && ww < WW)
        s += cw[d * 9 + (dy + 1) * 3 + (dx + 1)] * src[(long)(hh * WW + ww) * (2 * DI) + d];
    }
  float sig = 1.0f / (1.0f + __expf(-s));
  cbuf[idx] = s * sig;
}

// ---------------- materialize 4 scan directions -----------------------------
// XS layout: (k, 2*DI, LTOK); rows 0..143 = x branch, 144..287 = y branch
__global__ __launch_bounds__(256)
void dirs_kernel(const float* __restrict__ cbuf, float* __restrict__ XS) {
  int idx = blockIdx.x * blockDim.x + threadIdx.x;
  if (idx >= 2 * KDIR * DI * LTOK) return;
  int br  = idx / (KDIR * DI * LTOK);
  int rem = idx - br * KDIR * DI * LTOK;
  int k = rem / (DI * LTOK);
  int r2 = rem - k * DI * LTOK;
  int d = r2 / LTOK, l = r2 % LTOK;
  int srcl;
  if (k == 0)      srcl = l;
  else if (k == 1) srcl = (l % WW) * HH + l / WW;
  else if (k == 2) srcl = LTOK - 1 - l;
  else { int lf = LTOK - 1 - l; srcl = (lf % WW) * HH + lf / WW; }
  XS[(long)k * 2 * DI * LTOK + (long)(br * DI + d) * LTOK + l] =
      cbuf[(long)br * DI * LTOK + (long)d * LTOK + srcl];
}

// ---------------- interleave u / dts ---------------------------------------
__global__ __launch_bounds__(256)
void interleave_udts_kernel(const float* __restrict__ XS, const float* __restrict__ MARK,
                            const float* __restrict__ DTX, const float* __restrict__ DTY,
                            float* __restrict__ U, float* __restrict__ DTSB) {
  int idx = blockIdx.x * blockDim.x + threadIdx.x;
  if (idx >= KD * L2) return;
  int ch = idx / L2;               // k*DI + d
  int i  = idx % L2;
  int k = ch / DI, d = ch % DI;
  int br = i & 1, l = i >> 1;
  float m   = MARK[(long)ch * LTOK + l];
  float xsv = XS[(long)k * 2 * DI * LTOK + (long)(br * DI + d) * LTOK + l];
  U[idx] = xsv + m;
  const float* dt = br ? DTY : DTX;
  DTSB[idx] = dt[(long)ch * LTOK + l];
}

// ---------------- interleave B / C into (k, l2, n) lane-coalesced layout ----
__global__ __launch_bounds__(256)
void interleave_bc_kernel(const float* __restrict__ DBLX, const float* __restrict__ DBLY,
                          float* __restrict__ BS, float* __restrict__ CS) {
  int idx = blockIdx.x * blockDim.x + threadIdx.x;
  if (idx >= KDIR * L2 * DS) return;
  int n = idx % DS;
  int rest = idx / DS;
  int i = rest % L2;
  int k = rest / L2;
  int br = i & 1, l = i >> 1;
  const float* src = (br ? DBLY : DBLX) + (long)k * PROJP * LTOK;
  BS[idx] = src[(long)(DRNK + n) * LTOK + l];
  CS[idx] = src[(long)(DRNK + DS + n) * LTOK + l];
}

// ---------------- selective scan: 1 wave per channel, lane n = state n ------
__global__ __launch_bounds__(256)
void scan_kernel(const float* __restrict__ U, const float* __restrict__ DTSB,
                 const float* __restrict__ BS, const float* __restrict__ CS,
                 const float* __restrict__ A_logs, const float* __restrict__ Dsv,
                 const float* __restrict__ dt_bias,
                 float* __restrict__ OX, float* __restrict__ OY) {
  const int lane = threadIdx.x;
  const int ch = blockIdx.x * blockDim.y + threadIdx.y;    // wave-uniform
  if (ch >= KD) return;
  const int k = ch / DI;

  const float An   = -__expf(A_logs[(long)ch * DS + lane]);  // lane n: A[ch][n]
  const float Dv   = Dsv[ch];
  const float bias = dt_bias[ch];

  const float* ub = U    + (long)ch * L2;
  const float* db = DTSB + (long)ch * L2;
  const float* Bb = BS   + (long)k * L2 * DS;
  const float* Cb = CS   + (long)k * L2 * DS;
  float* ox = OX + (long)ch * LTOK;
  float* oy = OY + (long)ch * LTOK;

  float h = 0.0f;
  for (int base = 0; base < L2; base += 32) {
    float ur = ub[base + lane];       // coalesced 32-step staging
    float dr = db[base + lane];
    if (base + 32 < L2) {             // hint next B/C block into cache
      __builtin_prefetch(&Bb[(long)(base + 32) * DS + lane], 0, 0);
      __builtin_prefetch(&Cb[(long)(base + 32) * DS + lane], 0, 0);
    }
    float keep = 0.0f;
    for (int j = 0; j < 32; ++j) {
      float uu = __shfl(ur, j, 32);
      float dd = __shfl(dr, j, 32);
      float xs = dd + bias;
      float delta = (xs > 20.0f) ? xs : __logf(1.0f + __expf(xs));   // softplus
      float dA = __expf(delta * An);
      float bn = Bb[(long)(base + j) * DS + lane];
      float cn = Cb[(long)(base + j) * DS + lane];
      h = dA * h + delta * uu * bn;                                   // SSM step
      float y = h * cn;                                               // C proj
      y += __shfl_xor(y, 16, 32);
      y += __shfl_xor(y, 8, 32);
      y += __shfl_xor(y, 4, 32);
      y += __shfl_xor(y, 2, 32);
      y += __shfl_xor(y, 1, 32);
      if (lane == j) keep = y + Dv * uu;
    }
    int i = base + lane;              // parity(i) == parity(lane)
    if ((i & 1) == 0) ox[i >> 1] = keep; else oy[i >> 1] = keep;
  }
}

// ---------------- cross-merge the 4 directions ------------------------------
__global__ __launch_bounds__(256)
void recover_kernel(const float* __restrict__ OX, const float* __restrict__ OY,
                    float* __restrict__ XO, float* __restrict__ YO) {
  int idx = blockIdx.x * blockDim.x + threadIdx.x;
  if (idx >= 2 * DI * LTOK) return;
  int br  = idx / (DI * LTOK);
  int rem = idx - br * DI * LTOK;
  int d = rem / LTOK, l = rem % LTOK;
  int h = l / WW, w = l % WW;
  const float* o = br ? OY : OX;
  int lt = w * HH + h;
  float v = o[(long)(0 * DI + d) * LTOK + l]
          + o[(long)(2 * DI + d) * LTOK + (LTOK - 1 - l)]
          + o[(long)(1 * DI + d) * LTOK + lt]
          + o[(long)(3 * DI + d) * LTOK + (LTOK - 1 - lt)];
  (br ? YO : XO)[(long)d * LTOK + l] = v;
}

// ---------------- LayerNorm + SiLU gate, 1 wave per token -------------------
// writes into TB with row stride KPAD_DI (=160); pad cols are killed by the
// zero-padded out-proj weight, so they need no initialization.
__global__ __launch_bounds__(256)
void ln_gate_kernel(const float* __restrict__ xo, const float* __restrict__ zsrc,
                    const float* __restrict__ gamma, const float* __restrict__ beta,
                    float* __restrict__ tb) {
  const int lane = threadIdx.x;
  const int tok = blockIdx.x * blockDim.y + threadIdx.y;   // wave-uniform
  if (tok >= LTOK) return;
  float vals[5];
  float s = 0.0f;
#pragma unroll
  for (int ii = 0; ii < 5; ++ii) {
    int d = lane + ii * 32;
    vals[ii] = (d < DI) ? xo[(long)d * LTOK + tok] : 0.0f;
    s += vals[ii];
  }
  s += __shfl_xor(s, 16, 32); s += __shfl_xor(s, 8, 32);
  s += __shfl_xor(s, 4, 32);  s += __shfl_xor(s, 2, 32); s += __shfl_xor(s, 1, 32);
  float mu = s * (1.0f / DI);
  float v2 = 0.0f;
#pragma unroll
  for (int ii = 0; ii < 5; ++ii) {
    int d = lane + ii * 32;
    if (d < DI) { float t = vals[ii] - mu; v2 += t * t; }
  }
  v2 += __shfl_xor(v2, 16, 32); v2 += __shfl_xor(v2, 8, 32);
  v2 += __shfl_xor(v2, 4, 32);  v2 += __shfl_xor(v2, 2, 32); v2 += __shfl_xor(v2, 1, 32);
  float rstd = rsqrtf(v2 * (1.0f / DI) + 1e-5f);
#pragma unroll
  for (int ii = 0; ii < 5; ++ii) {
    int d = lane + ii * 32;
    if (d < DI) {
      float z = zsrc[(long)tok * (2 * DI) + DI + d];
      float sig = 1.0f / (1.0f + __expf(-z));
      float t = (vals[ii] - mu) * rstd * gamma[d] + beta[d];
      tb[(long)tok * KPAD_DI + d] = t * (z * sig);
    }
  }
}

// ---------------------------------------------------------------------------
// host-side launcher
// ---------------------------------------------------------------------------
static inline int gemm_blocks(int tilesM, int tilesN) {
  return (tilesM * tilesN + 7) / 8;
}
static inline int ew_blocks(long n) { return (int)((n + 255) / 256); }

extern "C" void kernel_launch(void* const* d_in, const int* in_sizes, int n_in,
                              void* d_out, int out_size, void* d_ws, size_t ws_size,
                              hipStream_t stream) {
  (void)in_sizes; (void)n_in; (void)out_size; (void)ws_size;

  const float* x      = (const float*)d_in[0];
  const float* y      = (const float*)d_in[1];
  const float* inpw   = (const float*)d_in[2];   // (288, 96)
  const float* convw  = (const float*)d_in[3];   // (144,1,3,3)
  const float* convb  = (const float*)d_in[4];   // (144)
  const float* xpw    = (const float*)d_in[5];   // (4,70,144)
  const float* ypw    = (const float*)d_in[6];   // (4,70,144)
  const float* dtw    = (const float*)d_in[7];   // (4,144,6)
  const float* dtb    = (const float*)d_in[8];   // (4,144)
  const float* alogs  = (const float*)d_in[9];   // (576,32)
  const float* dsv    = (const float*)d_in[10];  // (576)
  const float* markw  = (const float*)d_in[11];  // (144,288)
  const float* lng    = (const float*)d_in[12];
  const float* lnb    = (const float*)d_in[13];
  const float* outw   = (const float*)d_in[14];  // (96,144)
  float* out = (float*)d_out;

  // -------- workspace layout (float offsets) --------
  float* ws = (float*)d_ws;
  float* XZ    = ws;                                        // 2304*288
  float* YZ    = XZ    + (long)LTOK * 2 * DI;
  float* CBUF  = YZ    + (long)LTOK * 2 * DI;               // 2*144*2304
  float* XS    = CBUF  + (long)2 * DI * LTOK;               // 4*288*2304 + 16*2304 spare
  float* DBLX  = XS    + (long)KDIR * 2 * DI * LTOK + 16 * LTOK;  // 4*80*2304
  float* DBLY  = DBLX  + (long)KDIR * PROJP * LTOK;
  float* DTX   = DBLY  + (long)KDIR * PROJP * LTOK;         // 4*144*2304
  float* DTY   = DTX   + (long)KDIR * DI * LTOK;
  float* MARK  = DTY   + (long)KDIR * DI * LTOK;            // 4*144*2304
  float* Ub    = MARK  + (long)KDIR * DI * LTOK;            // 576*4608
  float* DTSB  = Ub    + (long)KD * L2;
  float* BSb   = DTSB  + (long)KD * L2;                     // 4*4608*32
  float* CSb   = BSb   + (long)KDIR * L2 * DS;
  float* OXb   = CSb   + (long)KDIR * L2 * DS;              // 576*2304
  float* OYb   = OXb   + (long)KD * LTOK;
  float* XO    = OYb   + (long)KD * LTOK;                   // 144*2304
  float* YO    = XO    + (long)DI * LTOK;
  float* TBX   = YO    + (long)DI * LTOK;                   // 2304*160 (pad cols uninit, ok)
  float* TBY   = TBX   + (long)LTOK * KPAD_DI;
  float* XPWP  = TBY   + (long)LTOK * KPAD_DI;              // 4*80*160 zero-padded
  float* YPWP  = XPWP  + (long)KDIR * PROJP * KPAD_DI;
  float* DTWP  = YPWP  + (long)KDIR * PROJP * KPAD_DI;      // 4*144*32 zero-padded
  float* OUTWP = DTWP  + (long)KDIR * DI * KPAD_DT;         // 96*160 zero-padded

  const dim3 wblk(32, 8);

  // 0) zero-padded weight copies (make all GEMM loads unconditional)
  padcopy_kernel<<<ew_blocks((long)KDIR * PROJP * KPAD_DI), 256, 0, stream>>>(
      xpw, XPWP, PROJ, DI, PROJP, KPAD_DI, KDIR);
  padcopy_kernel<<<ew_blocks((long)KDIR * PROJP * KPAD_DI), 256, 0, stream>>>(
      ypw, YPWP, PROJ, DI, PROJP, KPAD_DI, KDIR);
  padcopy_kernel<<<ew_blocks((long)KDIR * DI * KPAD_DT), 256, 0, stream>>>(
      dtw, DTWP, DI, DRNK, DI, KPAD_DT, KDIR);
  padcopy_kernel<<<ew_blocks((long)DM * KPAD_DI), 256, 0, stream>>>(
      outw, OUTWP, DM, DI, DM, KPAD_DI, 1);

  // 1) in_proj: xz = x @ W^T  (M=2304, N=288, K=96 -> 3 steps; B contiguous)
  wmma_gemm_bf16<3, 1><<<dim3(gemm_blocks(144, 18), 1), wblk, 0, stream>>>(
      x, inpw, XZ, 18, 144 * 18, DM, DM, 2 * DI, 0, 0, 0);
  wmma_gemm_bf16<3, 1><<<dim3(gemm_blocks(144, 18), 1), wblk, 0, stream>>>(
      y, inpw, YZ, 18, 144 * 18, DM, DM, 2 * DI, 0, 0, 0);

  // 2) depthwise conv + SiLU
  conv_silu_kernel<<<ew_blocks((long)2 * DI * LTOK), 256, 0, stream>>>(
      XZ, YZ, convw, convb, CBUF);

  // 3) four scan directions
  dirs_kernel<<<ew_blocks((long)2 * KDIR * DI * LTOK), 256, 0, stream>>>(CBUF, XS);

  // 4) x_proj / y_proj: (80x160 padded)@(144x2304), batch=4, K=160 -> 5 steps
  wmma_gemm_bf16<5, LTOK><<<dim3(gemm_blocks(5, 144), KDIR), wblk, 0, stream>>>(
      XPWP, XS, DBLX, 144, 5 * 144, KPAD_DI, 1, LTOK,
      (long)PROJP * KPAD_DI, (long)2 * DI * LTOK, (long)PROJP * LTOK);
  wmma_gemm_bf16<5, LTOK><<<dim3(gemm_blocks(5, 144), KDIR), wblk, 0, stream>>>(
      YPWP, XS + (long)DI * LTOK, DBLY, 144, 5 * 144, KPAD_DI, 1, LTOK,
      (long)PROJP * KPAD_DI, (long)2 * DI * LTOK, (long)PROJP * LTOK);

  // 5) dt_proj: (144x32 padded)@(rows 0..5 of dbl), batch=4, K=32 -> 1 step
  wmma_gemm_bf16<1, LTOK><<<dim3(gemm_blocks(9, 144), KDIR), wblk, 0, stream>>>(
      DTWP, DBLX, DTX, 144, 9 * 144, KPAD_DT, 1, LTOK,
      (long)DI * KPAD_DT, (long)PROJP * LTOK, (long)DI * LTOK);
  wmma_gemm_bf16<1, LTOK><<<dim3(gemm_blocks(9, 144), KDIR), wblk, 0, stream>>>(
      DTWP, DBLY, DTY, 144, 9 * 144, KPAD_DT, 1, LTOK,
      (long)DI * KPAD_DT, (long)PROJP * LTOK, (long)DI * LTOK);

  // 6) fusion mark: (144x288)@(288x2304), batch=4 (shared weights), 9 steps
  wmma_gemm_bf16<9, LTOK><<<dim3(gemm_blocks(9, 144), KDIR), wblk, 0, stream>>>(
      markw, XS, MARK, 144, 9 * 144, 2 * DI, 1, LTOK,
      0, (long)2 * DI * LTOK, (long)DI * LTOK);

  // 7) interleave x/y tokens -> u, dts, Bs, Cs
  interleave_udts_kernel<<<ew_blocks((long)KD * L2), 256, 0, stream>>>(
      XS, MARK, DTX, DTY, Ub, DTSB);
  interleave_bc_kernel<<<ew_blocks((long)KDIR * L2 * DS), 256, 0, stream>>>(
      DBLX, DBLY, BSb, CSb);

  // 8) selective scan: 576 waves, 8 waves/block
  scan_kernel<<<dim3((KD + 7) / 8), wblk, 0, stream>>>(
      Ub, DTSB, BSb, CSb, alogs, dsv, dtb, OXb, OYb);

  // 9) cross-merge directions
  recover_kernel<<<ew_blocks((long)2 * DI * LTOK), 256, 0, stream>>>(OXb, OYb, XO, YO);

  // 10) LayerNorm + gate (writes TB with row stride 160)
  ln_gate_kernel<<<dim3((LTOK + 7) / 8), wblk, 0, stream>>>(XO, XZ, lng, lnb, TBX);
  ln_gate_kernel<<<dim3((LTOK + 7) / 8), wblk, 0, stream>>>(YO, YZ, lng, lnb, TBY);

  // 11) out_proj: (2304x160 padded-A)@(160x96 zero-padded-B), K=160, B contig
  wmma_gemm_bf16<5, 1><<<dim3(gemm_blocks(144, 6), 1), wblk, 0, stream>>>(
      TBX, OUTWP, out, 6, 144 * 6, KPAD_DI, KPAD_DI, DM, 0, 0, 0);
  wmma_gemm_bf16<5, 1><<<dim3(gemm_blocks(144, 6), 1), wblk, 0, stream>>>(
      TBY, OUTWP, out + (long)LTOK * DM, 6, 144 * 6, KPAD_DI, KPAD_DI, DM, 0, 0, 0);
}
